// SlotVectorQuantizer_2869038154047
// MI455X (gfx1250) — compile-verified, matrix operand order checked by reading, not run on hardware
//
#include <hip/hip_runtime.h>
#include <hip/hip_bf16.h>

typedef __attribute__((ext_vector_type(16))) __bf16 v16bf;
typedef __attribute__((ext_vector_type(8)))  __bf16 v8bf;
typedef __attribute__((ext_vector_type(8)))  float  v8f;

#define B_   16
#define N_   256
#define EDIM 1024
#define F_   2
#define C_   512
#define NE   4096
#define NCT  (NE/16)                 /* 256 column tiles per codebook      */
#define ROWS_PER_F (B_*N_)           /* 4096 rows per codebook             */
#define TOTAL_ROWS (F_*ROWS_PER_F)   /* 8192                               */
#define ZQ_ELEMS   (B_*N_*EDIM)      /* 4194304 floats of z_q_out          */
#define PITCH_BF   520               /* LDS row pitch in bf16 (1040 B):    */
                                     /* 260 dwords -> lane stride 4 banks, */
                                     /* 16 lanes tile all 64 banks         */

// ---------------------------------------------------------------------------
// Kernel 1/2: L2-normalize 512-elem rows, emit bf16.
// zmode=1: input is z (b,n,1024); output row r = f*4096 + (b*256+n) pulls
//          the f-th 512-chunk of input row (b*256+n).
// zmode=0: input is emb_weight (8192,512), plain row-major.
// ---------------------------------------------------------------------------
__global__ void rownorm_to_bf16(const float* __restrict__ in,
                                __bf16* __restrict__ out, int zmode)
{
    const int r = blockIdx.x;         // 0..8191
    const int t = threadIdx.x;        // 0..255
    const float* src;
    if (zmode) {
        const int f  = r >> 12;       // / 4096
        const int bn = r & 4095;
        src = in + (size_t)bn * EDIM + (size_t)f * C_;
    } else {
        src = in + (size_t)r * C_;
    }
    const float x0 = src[t];
    const float x1 = src[t + 256];

    __shared__ float red[256];
    red[t] = x0 * x0 + x1 * x1;
    __syncthreads();
    for (int o = 128; o > 0; o >>= 1) {
        if (t < o) red[t] += red[t + o];
        __syncthreads();
    }
    const float scale = 1.0f / fmaxf(sqrtf(red[0]), 1e-12f);

    __bf16* dst = out + (size_t)r * C_;
    dst[t]       = (__bf16)(x0 * scale);
    dst[t + 256] = (__bf16)(x1 * scale);
}

// ---------------------------------------------------------------------------
// Stage one 16-column B tile (16 rows x 1024 B) into padded LDS.
// 256 threads x 4 x 16-byte chunks. global_load_b128 -> ds_store_b128.
// ---------------------------------------------------------------------------
__device__ __forceinline__ void stageB(const __bf16* __restrict__ pbase,
                                       int ct, __bf16* dst, int t)
{
#pragma unroll
    for (int i = 0; i < 4; ++i) {
        const int q     = t + i * 256;   // 0..1023 chunk id
        const int row   = q >> 6;        // 0..15 (column within tile)
        const int off16 = q & 63;        // 16B chunk within the 1KB row
        const uint4 v = *(const uint4*)(pbase +
                          (size_t)(ct * 16 + row) * C_ + off16 * 8);
        *(uint4*)(dst + row * PITCH_BF + off16 * 8) = v;
    }
}

// ---------------------------------------------------------------------------
// Kernel 3: cosine-sim GEMM + row argmax via v_wmma_f32_16x16x32_bf16.
// 256-thread blocks = 8 waves, each wave one 16-row tile; all 8 waves share
// the same codebook so each B tile is fetched from L2 ONCE per block and
// broadcast through LDS (8x L2-traffic reduction), double-buffered.
// A-fragments (16 rows x K=512) stay register-resident per wave.
// B fragments are software-pipelined one k-step ahead so each WMMA's LDS
// loads were issued a full iteration earlier (relaxes s_wait_dscnt 0x0).
// 512 row tiles = 64 blocks x 8 waves. EXEC all-ones around every WMMA.
// ---------------------------------------------------------------------------
__global__ __launch_bounds__(256)
void argmax_wmma(const __bf16* __restrict__ zn,
                 const __bf16* __restrict__ pn,
                 int* __restrict__ idx_out)
{
    __shared__ __align__(16) __bf16 ldsB[2][16 * PITCH_BF];

    const int t    = threadIdx.x;
    const int wave = t >> 5;
    const int ln   = t & 31;
    const int half = ln >> 4;         // lane group (K-offset select)
    const int lnm  = ln & 15;         // M (for A) / N (for B) within tile

    const int rt      = blockIdx.x * 8 + wave;  // 0..511
    const int f       = blockIdx.x >> 5;        // codebook (uniform per block)
    const int rowBase = (rt & 255) << 4;        // 16-row tile origin

    // --- preload A: 16 fragments covering K = 0..511 for this row tile ---
    const __bf16* arow =
        zn + ((size_t)f * ROWS_PER_F + rowBase + lnm) * C_ + half * 8;
    v16bf afrag[16];
#pragma unroll
    for (int ks = 0; ks < 16; ++ks) {
        const v8bf lo = *(const v8bf*)(arow + ks * 32);
        const v8bf hi = *(const v8bf*)(arow + ks * 32 + 16);
        v16bf fr;
#pragma unroll
        for (int i = 0; i < 8; ++i) { fr[i] = lo[i]; fr[8 + i] = hi[i]; }
        afrag[ks] = fr;
    }

    float bestv[8];
    int   besti[8];
#pragma unroll
    for (int v = 0; v < 8; ++v) { bestv[v] = -3.0e30f; besti[v] = 0; }

    const __bf16* pbase = pn + (size_t)f * NE * C_;

    stageB(pbase, 0, &ldsB[0][0], t);

    for (int ct = 0; ct < NCT; ++ct) {
        __syncthreads();   // buf[ct&1] staged; buf[(ct+1)&1] free to refill
        if (ct + 2 < NCT)  // pull tile ct+2 toward L2 -> global_prefetch_b8
            __builtin_prefetch(pbase + (size_t)(ct + 2) * 16 * C_ + t * 32, 0, 0);
        if (ct + 1 < NCT)
            stageB(pbase, ct + 1, &ldsB[(ct + 1) & 1][0], t);

        // compute: B fragments from LDS (ds_load_b128, bank-conflict-free),
        // pipelined one k-step ahead of the consuming WMMA.
        const __bf16* brow = &ldsB[ct & 1][lnm * PITCH_BF + half * 8];
        v8bf lo = *(const v8bf*)(brow);
        v8bf hi = *(const v8bf*)(brow + 16);
        v8f acc = {};
#pragma unroll
        for (int ks = 0; ks < 16; ++ks) {
            v8bf nlo = lo, nhi = hi;
            if (ks + 1 < 16) {
                nlo = *(const v8bf*)(brow + (ks + 1) * 32);
                nhi = *(const v8bf*)(brow + (ks + 1) * 32 + 16);
            }
            v16bf bf;
#pragma unroll
            for (int i = 0; i < 8; ++i) { bf[i] = lo[i]; bf[8 + i] = hi[i]; }
            acc = __builtin_amdgcn_wmma_f32_16x16x32_bf16(
                false, afrag[ks], false, bf, (short)0, acc, false, false);
            lo = nlo; hi = nhi;
        }

        const int col = ct * 16 + lnm;
#pragma unroll
        for (int v = 0; v < 8; ++v) {
            const float s = acc[v];
            if (s > bestv[v]) { bestv[v] = s; besti[v] = col; }
        }
    }

    // C/D layout: VGPR v holds row M = v + 8*half, column N = lnm.
    // Reduce (max, argmin-on-tie) across the 16 lanes of each half.
#pragma unroll
    for (int v = 0; v < 8; ++v) {
        float bv = bestv[v];
        int   bi = besti[v];
#pragma unroll
        for (int m = 1; m < 16; m <<= 1) {
            const float ov = __shfl_xor(bv, m, 16);
            const int   oi = __shfl_xor(bi, m, 16);
            if (ov > bv || (ov == bv && oi < bi)) { bv = ov; bi = oi; }
        }
        if (lnm == 0) {
            const int M = v + half * 8;
            idx_out[(size_t)f * ROWS_PER_F + rowBase + M] = bi;
        }
    }
}

// ---------------------------------------------------------------------------
// Kernel 4: gather emb_weight[idx] from the FULL table (faithful: no codebook
// offset), L2-normalize in fp32, write z_q_out, accumulate loss, emit index.
// ---------------------------------------------------------------------------
__global__ void gather_out(const float* __restrict__ z,
                           const float* __restrict__ emb,
                           const int* __restrict__ idx_ws,
                           float* __restrict__ out,
                           float* __restrict__ lossAcc)
{
    const int r   = blockIdx.x;        // b*512 + f*256 + n  (output index order)
    const int n_i = r & 255;
    const int f_i = (r >> 8) & 1;
    const int b_i = r >> 9;
    const int t   = threadIdx.x;

    const int bn  = b_i * N_ + n_i;
    const int idx = idx_ws[f_i * ROWS_PER_F + bn];

    const float* erow = emb + (size_t)idx * C_;
    const float* zrow = z + (size_t)bn * EDIM + (size_t)f_i * C_;

    const float e0 = erow[t], e1 = erow[t + 256];
    const float z0 = zrow[t], z1 = zrow[t + 256];

    __shared__ float redE[256], redZ[256];
    redE[t] = e0 * e0 + e1 * e1;
    redZ[t] = z0 * z0 + z1 * z1;
    __syncthreads();
    for (int o = 128; o > 0; o >>= 1) {
        if (t < o) { redE[t] += redE[t + o]; redZ[t] += redZ[t + o]; }
        __syncthreads();
    }
    const float se = 1.0f / fmaxf(sqrtf(redE[0]), 1e-12f);
    const float sz = 1.0f / fmaxf(sqrtf(redZ[0]), 1e-12f);

    const float q0 = e0 * se, q1 = e1 * se;
    const float d0 = q0 - z0 * sz, d1 = q1 - z1 * sz;

    float* orow = out + (size_t)bn * EDIM + (size_t)f_i * C_;
    orow[t]       = q0;   // straight-through output == z_q numerically
    orow[t + 256] = q1;

    __syncthreads();
    redE[t] = d0 * d0 + d1 * d1;
    __syncthreads();
    for (int o = 128; o > 0; o >>= 1) {
        if (t < o) redE[t] += redE[t + o];
        __syncthreads();
    }
    if (t == 0) {
        atomicAdd(lossAcc, redE[0]);
        out[ZQ_ELEMS + 1 + r] = (float)idx;   // encoding_indices (b,f,n)
    }
}

__global__ void finalize_loss(const float* __restrict__ lossAcc,
                              float* __restrict__ out)
{
    // 0.25*mean + mean == 1.25*mean (stop_gradient is identity in value)
    out[ZQ_ELEMS] = 1.25f * lossAcc[0] / (float)ZQ_ELEMS;
}

// ---------------------------------------------------------------------------
extern "C" void kernel_launch(void* const* d_in, const int* in_sizes, int n_in,
                              void* d_out, int out_size, void* d_ws, size_t ws_size,
                              hipStream_t stream)
{
    const float* z   = (const float*)d_in[0];   // (16,256,1024) f32
    const float* emb = (const float*)d_in[1];   // (8192,512)    f32
    float* out = (float*)d_out;

    char* ws = (char*)d_ws;
    const size_t bf16RowsBytes = (size_t)TOTAL_ROWS * C_ * sizeof(__bf16); // 8 MB
    __bf16* zn_bf   = (__bf16*)ws;
    __bf16* pn_bf   = (__bf16*)(ws + bf16RowsBytes);
    int*    idx_ws  = (int*)  (ws + 2 * bf16RowsBytes);
    float*  lossAcc = (float*)(ws + 2 * bf16RowsBytes + (size_t)TOTAL_ROWS * sizeof(int));

    hipMemsetAsync(lossAcc, 0, sizeof(float), stream);

    rownorm_to_bf16<<<TOTAL_ROWS, 256, 0, stream>>>(z,   zn_bf, 1);
    rownorm_to_bf16<<<TOTAL_ROWS, 256, 0, stream>>>(emb, pn_bf, 0);

    // 512 row tiles (2 codebooks x 256), 8 waves per block sharing B via LDS
    argmax_wmma<<<64, 256, 0, stream>>>(zn_bf, pn_bf, idx_ws);

    gather_out<<<TOTAL_ROWS, 256, 0, stream>>>(z, emb, idx_ws, out, lossAcc);
    finalize_loss<<<1, 1, 0, stream>>>(lossAcc, out);
}
